// Head_13237089206941
// MI455X (gfx1250) — matrix operand order repaired and checked by reading, not compiled, not run on hardware
//
#include <hip/hip_runtime.h>
#include <math.h>

typedef __attribute__((ext_vector_type(16))) _Float16 v16h;
typedef __attribute__((ext_vector_type(8)))  _Float16 v8h;
typedef __attribute__((ext_vector_type(8)))  float    v8f;

#define HS 64
#define NEG_BIG (-3.402823466e38f)
// reference multiplies logits by C^0.5 = 8; fold log2(e) too so softmax is
// raw v_exp_f32 (exp2) with no per-element multiply.
#define Q_SCALE (8.0f * 1.4426950408889634f)

// ---------------------------------------------------------------------------
// Kernel 1: projections.  Q = (8*log2e) * x@Wq (f16), K = x@Wk (f16, row
// major), Vt = (x@Wv)^T stored [B][HS][T] (f16) so P@V B-fragments are
// contiguous 32B loads per lane.
// ---------------------------------------------------------------------------
__global__ __launch_bounds__(256)
void proj_kernel(const float* __restrict__ x,
                 const float* __restrict__ Wq,
                 const float* __restrict__ Wk,
                 const float* __restrict__ Wv,
                 _Float16* __restrict__ Q,
                 _Float16* __restrict__ Kk,
                 _Float16* __restrict__ Vt,
                 int T)
{
    __shared__ float xs[256];                       // 4 rows x 64
    const int tid = threadIdx.x;
    xs[tid] = x[(size_t)blockIdx.x * 256 + tid];
    __syncthreads();

    const int r = tid >> 6;                         // local row 0..3
    const int h = tid & 63;                         // output column
    float aq = 0.f, ak = 0.f, av = 0.f;
    #pragma unroll
    for (int c = 0; c < 64; ++c) {
        const float xv = xs[r * 64 + c];
        aq += xv * Wq[c * 64 + h];
        ak += xv * Wk[c * 64 + h];
        av += xv * Wv[c * 64 + h];
    }
    const size_t bt = (size_t)blockIdx.x * 4 + r;   // global b*T + t
    Q [bt * HS + h] = (_Float16)(aq * Q_SCALE);
    Kk[bt * HS + h] = (_Float16)ak;
    const size_t b = bt / (size_t)T, t = bt % (size_t)T;
    Vt[(b * HS + h) * (size_t)T + t] = (_Float16)av;
}

// ---------------------------------------------------------------------------
// Kernel 2: flash attention.  One wave owns 16 query rows; 4 waves per block.
// KV tiles of 64 (4 N-subtiles): 16 v_wmma per tile, one softmax reduction
// pass.  The diagonal tile is PEELED: the main loop is straight-line with no
// mask code and no branches; only the final tile applies the causal mask.
// Online softmax in exp2 domain, D-layout; P transposed to A-layout through
// wave-private LDS.  Scalar (readfirstlane) control flow keeps EXEC all-1s
// at every WMMA.
// ---------------------------------------------------------------------------
__global__ __launch_bounds__(128)
void attn_kernel(const _Float16* __restrict__ Q,
                 const _Float16* __restrict__ Kk,
                 const _Float16* __restrict__ Vt,
                 float* __restrict__ out,
                 int T)
{
    const int warp = __builtin_amdgcn_readfirstlane((int)(threadIdx.x >> 5));
    const int lane = threadIdx.x & 31;
    const int l16  = lane & 15;
    const int hlf  = lane >> 4;                     // lane half (0/1)
    const int b    = blockIdx.y;
    const int q0   = __builtin_amdgcn_readfirstlane(blockIdx.x * 64 + warp * 16);

    __shared__ __align__(32) _Float16 pbuf[4][16 * 64];
    _Float16* pw = pbuf[warp];                      // wave-private 16x64 strip

    // --- preload Q A-fragments (2 chunks of K=32), register resident -------
    const _Float16* qrow = Q + ((size_t)b * T + (q0 + l16)) * HS;
    v16h qa[2];
    #pragma unroll
    for (int c = 0; c < 2; ++c) {
        const int kk = c * 32;
        v8h lo = *(const v8h*)(qrow + kk + 16 * hlf);       // K kk+{0..7 | 16..23}
        v8h hi = *(const v8h*)(qrow + kk + 8 + 16 * hlf);   // K kk+{8..15 | 24..31}
        qa[c] = __builtin_shufflevector(lo, hi, 0,1,2,3,4,5,6,7,
                                                 8,9,10,11,12,13,14,15);
    }

    v8f acc[4];                                     // O strip: 16 x 64 f32
    float m[8], l[8];
    #pragma unroll
    for (int t = 0; t < 4; ++t) acc[t] = (v8f){0.f,0.f,0.f,0.f,0.f,0.f,0.f,0.f};
    #pragma unroll
    for (int i = 0; i < 8; ++i) { m[i] = NEG_BIG; l[i] = 0.f; }

    // one KV-tile step; domask is a literal at each call site -> folds away
    auto kv_step = [&](const int kv, const bool domask) __attribute__((always_inline)) {
        // ---- S = Q . K^T : 4 N-subtiles x 2 K-chunks ----------------------
        v8f s[4];
        #pragma unroll
        for (int j = 0; j < 4; ++j) {
            s[j] = (v8f){0.f,0.f,0.f,0.f,0.f,0.f,0.f,0.f};
            #pragma unroll
            for (int c = 0; c < 2; ++c) {
                v16h bk = *(const v16h*)(Kk + ((size_t)b*T + (kv + 16*j + l16))*HS
                                            + c*32 + 16*hlf);
                s[j] = __builtin_amdgcn_wmma_f32_16x16x32_f16(
                           false, qa[c], false, bk, (short)0, s[j], false, false);
            }
        }

        // ---- causal mask (diagonal tile only; compile-time selected) ------
        if (domask) {
            #pragma unroll
            for (int j = 0; j < 4; ++j)
                #pragma unroll
                for (int i = 0; i < 8; ++i)
                    if (kv + 16*j + l16 > q0 + i + 8*hlf) s[j][i] = NEG_BIG;
        }

        // ---- online softmax, exp2 domain; one reduction per 64 columns ----
        #pragma unroll
        for (int i = 0; i < 8; ++i) {
            float mx = fmaxf(fmaxf(s[0][i], s[1][i]), fmaxf(s[2][i], s[3][i]));
            mx = fmaxf(mx, __shfl_xor(mx, 8, 32));
            mx = fmaxf(mx, __shfl_xor(mx, 4, 32));
            mx = fmaxf(mx, __shfl_xor(mx, 2, 32));
            mx = fmaxf(mx, __shfl_xor(mx, 1, 32));
            const float mn    = fmaxf(m[i], mx);
            const float alpha = exp2f(m[i] - mn);
            const float p0 = exp2f(s[0][i] - mn);
            const float p1 = exp2f(s[1][i] - mn);
            const float p2 = exp2f(s[2][i] - mn);
            const float p3 = exp2f(s[3][i] - mn);
            float rs = (p0 + p1) + (p2 + p3);
            rs += __shfl_xor(rs, 8, 32);
            rs += __shfl_xor(rs, 4, 32);
            rs += __shfl_xor(rs, 2, 32);
            rs += __shfl_xor(rs, 1, 32);
            l[i] = l[i] * alpha + rs;
            m[i] = mn;
            acc[0][i] *= alpha; acc[1][i] *= alpha;
            acc[2][i] *= alpha; acc[3][i] *= alpha;
            // transpose P (D-layout -> row-major strip) through LDS
            const int prow = i + 8 * hlf;
            pw[prow * 64 +      l16] = (_Float16)p0;
            pw[prow * 64 + 16 + l16] = (_Float16)p1;
            pw[prow * 64 + 32 + l16] = (_Float16)p2;
            pw[prow * 64 + 48 + l16] = (_Float16)p3;
        }

        // same-wave DS ordering; make it explicit + fence the compiler
        asm volatile("s_wait_dscnt 0" ::: "memory");

        // ---- O += P . V : 2 K-chunks x 4 N-subtiles -----------------------
        #pragma unroll
        for (int c = 0; c < 2; ++c) {
            v8h plo = *(const v8h*)(pw + l16 * 64 + c*32 + 16 * hlf);
            v8h phi = *(const v8h*)(pw + l16 * 64 + c*32 + 8 + 16 * hlf);
            v16h pa = __builtin_shufflevector(plo, phi, 0,1,2,3,4,5,6,7,
                                                        8,9,10,11,12,13,14,15);
            #pragma unroll
            for (int t = 0; t < 4; ++t) {
                v16h bv = *(const v16h*)(Vt + ((size_t)b*HS + (t*16 + l16))*T
                                            + kv + c*32 + 16*hlf);
                acc[t] = __builtin_amdgcn_wmma_f32_16x16x32_f16(
                             false, pa, false, bv, (short)0, acc[t], false, false);
            }
        }
    };

    // main loop: all tiles strictly below the diagonal, branch-free body
    const int qlast   = q0 + 15;
    const int kv_last = __builtin_amdgcn_readfirstlane((qlast / 64) * 64);
    for (int kv = 0; kv < kv_last; kv += 64)
        kv_step(kv, false);
    // peeled diagonal tile with causal mask
    kv_step(kv_last, true);

    // ---- write O / l (fp32) ----------------------------------------------
    #pragma unroll
    for (int t = 0; t < 4; ++t) {
        #pragma unroll
        for (int i = 0; i < 8; ++i) {
            const int row = q0 + i + 8 * hlf;
            out[((size_t)b * T + row) * HS + t * 16 + l16] = acc[t][i] / l[i];
        }
    }
}

// ---------------------------------------------------------------------------
extern "C" void kernel_launch(void* const* d_in, const int* in_sizes, int n_in,
                              void* d_out, int out_size, void* d_ws, size_t ws_size,
                              hipStream_t stream) {
    const float* x  = (const float*)d_in[0];
    const float* Wq = (const float*)d_in[1];
    const float* Wk = (const float*)d_in[2];
    const float* Wv = (const float*)d_in[3];
    float* out = (float*)d_out;

    const int B = 4;
    const int T = in_sizes[0] / (B * HS);           // 4096

    const size_t mat = (size_t)B * T * HS;          // elements per f16 matrix
    _Float16* Q  = (_Float16*)d_ws;                 // [B][T][HS]
    _Float16* Kk = Q + mat;                         // [B][T][HS]
    _Float16* Vt = Kk + mat;                        // [B][HS][T]

    proj_kernel<<<(B * T) / 4, 256, 0, stream>>>(x, Wq, Wk, Wv, Q, Kk, Vt, T);

    dim3 grid(T / 64, B);
    attn_kernel<<<grid, 128, 0, stream>>>(Q, Kk, Vt, out, T);
}
